// Attention_75230647157330
// MI455X (gfx1250) — matrix operand order repaired and checked by reading, not compiled, not run on hardware
//
#include <hip/hip_runtime.h>
#include <hip/hip_bf16.h>
#include <math.h>

// Problem constants (from reference): B=64, T=2048, D=512
#define BN 64
#define TT 2048
#define DD 512
#define NC 16              // t-chunks for weighted-sum partials
#define TCHUNK (TT / NC)   // 128

typedef __attribute__((ext_vector_type(16))) _Float16 v16h;
typedef __attribute__((ext_vector_type(8)))  float    v8f;

// ---------------------------------------------------------------------------
// WMMA GEMM: Out[M x N] = X[M x K] @ (TRANSB ? W^T : W) + bias
//   TRANSB=1:  Out[m,n] = sum_k X[m,k] * W[n*K + k]   (torch Linear)
//   TRANSB=0:  Out[m,n] = sum_k X[m,k] * W[k*N + n]
// One wave per 16x16 output tile; f32 -> f16 convert on load, f32 accumulate.
// Per-lane layouts used here (wave32; A and B share contiguous k order):
//   A 16x32 f16 : lane = l + 16h, M = l; half j -> K = 16h + j
//   B 32x16 f16 : lane = l + 16h, N = l; half j -> K = 16h + j
//   C/D 16x16   : lane = l + 16h, N = l; VGPR v -> M = v + 8h
// ---------------------------------------------------------------------------
template <bool TRANSB>
__global__ void wmma_gemm_f32(const float* __restrict__ X,
                              const float* __restrict__ W,
                              const float* __restrict__ bias,
                              float* __restrict__ Out,
                              int M, int N, int K) {
  const int lane = threadIdx.x & 31;
  const int l    = lane & 15;
  const int h    = lane >> 4;
  const int n0   = blockIdx.x * 16;
  const int m0   = blockIdx.y * 16;

  v8f acc = {};
  for (int k0 = 0; k0 < K; k0 += 32) {
    v16h a, b;
    // A-matrix gather: row (m0+l) of X, K = k0 + 16h + j
    const float* xr = X + (size_t)(m0 + l) * K + k0 + 16 * h;
#pragma unroll
    for (int j = 0; j < 8; ++j) a[j]     = (_Float16)xr[j];       // K = k0+16h+j
#pragma unroll
    for (int j = 0; j < 8; ++j) a[8 + j] = (_Float16)xr[8 + j];   // K = k0+16h+8+j
    // B-matrix gather: column (n0+l), K = k0 + 16h + j
    if (TRANSB) {
      const float* wr = W + (size_t)(n0 + l) * K + k0 + 16 * h;
#pragma unroll
      for (int j = 0; j < 16; ++j) b[j] = (_Float16)wr[j];
    } else {
      const float* wc = W + (size_t)(k0 + 16 * h) * N + (n0 + l);
#pragma unroll
      for (int j = 0; j < 16; ++j) b[j] = (_Float16)wc[(size_t)j * N];
    }
    acc = __builtin_amdgcn_wmma_f32_16x16x32_f16(
        /*neg_a=*/false, a, /*neg_b=*/false, b,
        /*c_mod=*/(short)0, acc, /*reuse_a=*/false, /*reuse_b=*/false);
  }

  const float bn = bias ? bias[n0 + l] : 0.0f;
#pragma unroll
  for (int v = 0; v < 8; ++v) {
    const int m = m0 + v + 8 * h;
    Out[(size_t)m * N + (n0 + l)] = acc[v] + bn;
  }
}

// c[b] = q[b,:] . b_enc  (one wave per b)
__global__ void dot_bias_kernel(const float* __restrict__ q,
                                const float* __restrict__ b_enc,
                                float* __restrict__ cb) {
  const int b = blockIdx.x;
  const int lane = threadIdx.x;
  float s = 0.0f;
  for (int d = lane; d < DD; d += 32) s += q[(size_t)b * DD + d] * b_enc[d];
#pragma unroll
  for (int off = 16; off > 0; off >>= 1) s += __shfl_down(s, off, 32);
  if (lane == 0) cb[b] = s;
}

// energy[b,t] = LF[b,t,:] . qhat[b,:] + c[b]
// One wave per 4 consecutive t rows: qhat[b] (2KB) held in 16 VGPRs and
// reused across the 4 rows; LF rows streamed with b128 loads.
__global__ void energy_kernel(const float* __restrict__ LF,
                              const float* __restrict__ qhat,
                              const float* __restrict__ cb,
                              float* __restrict__ energy) {
  const int gwave = (int)((blockIdx.x * blockDim.x + threadIdx.x) >> 5);
  const int lane  = threadIdx.x & 31;
  const int b  = gwave >> 9;               // TT/4 = 512 groups per b
  const int t0 = (gwave & 511) * 4;

  const float* __restrict__ qh = qhat + (size_t)b * DD;
  float4 y[4];
#pragma unroll
  for (int c = 0; c < 4; ++c) y[c] = *(const float4*)(qh + c * 128 + lane * 4);
  const float cbv = cb[b];

#pragma unroll
  for (int r = 0; r < 4; ++r) {
    const float* __restrict__ row = LF + ((size_t)b * TT + t0 + r) * DD;
    float s = 0.0f;
#pragma unroll
    for (int c = 0; c < 4; ++c) {
      const float4 x = *(const float4*)(row + c * 128 + lane * 4);
      s += x.x * y[c].x + x.y * y[c].y + x.z * y[c].z + x.w * y[c].w;
    }
#pragma unroll
    for (int off = 16; off > 0; off >>= 1) s += __shfl_down(s, off, 32);
    if (lane == 0) energy[(size_t)b * TT + t0 + r] = s + cbv;
  }
}

// softmax over T per batch row (one 256-thread block per b)
__global__ void softmax_kernel(const float* __restrict__ energy,
                               float* __restrict__ attn) {
  __shared__ float red[256];
  const int b = blockIdx.x;
  const int tid = threadIdx.x;
  const float* __restrict__ e = energy + (size_t)b * TT;
  float vals[8];
  float m = -INFINITY;
#pragma unroll
  for (int i = 0; i < 8; ++i) { vals[i] = e[tid + 256 * i]; m = fmaxf(m, vals[i]); }
  red[tid] = m;
  __syncthreads();
  for (int s = 128; s > 0; s >>= 1) {
    if (tid < s) red[tid] = fmaxf(red[tid], red[tid + s]);
    __syncthreads();
  }
  m = red[0];
  __syncthreads();
  float sum = 0.0f;
#pragma unroll
  for (int i = 0; i < 8; ++i) { vals[i] = __expf(vals[i] - m); sum += vals[i]; }
  red[tid] = sum;
  __syncthreads();
  for (int s = 128; s > 0; s >>= 1) {
    if (tid < s) red[tid] += red[tid + s];
    __syncthreads();
  }
  const float inv = 1.0f / red[0];
#pragma unroll
  for (int i = 0; i < 8; ++i) attn[(size_t)b * TT + tid + 256 * i] = vals[i] * inv;
}

// partial[c][b][e] = sum_{t in chunk c} attn[b,t] * LF[b,t,e]
// grid (BN, NC), 128 threads; thread owns 4 consecutive columns (b128 loads),
// one lane per 128B line issues a global_prefetch_b8 16 rows ahead.
__global__ void weighted_sum_kernel(const float* __restrict__ LF,
                                    const float* __restrict__ attn,
                                    float* __restrict__ partial) {
  const int b  = blockIdx.x;
  const int c  = blockIdx.y;
  const int e4 = threadIdx.x * 4;
  const int t0 = c * TCHUNK;
  const float* __restrict__ base = LF + ((size_t)b * TT + t0) * DD + e4;
  const float* __restrict__ a    = attn + (size_t)b * TT + t0;
  float4 acc = {0.0f, 0.0f, 0.0f, 0.0f};
  for (int t = 0; t < TCHUNK; ++t) {
    if ((threadIdx.x & 7) == 0)   // speculative prefetch; drops past end of LF
      __builtin_prefetch(base + (size_t)(t + 16) * DD, 0, 0);
    const float  at = a[t];
    const float4 x  = *(const float4*)(base + (size_t)t * DD);
    acc.x += at * x.x;
    acc.y += at * x.y;
    acc.z += at * x.z;
    acc.w += at * x.w;
  }
  *(float4*)(partial + ((size_t)c * BN + b) * DD + e4) = acc;
}

// v[b,e] = sum_c partial[c][b][e]   (deterministic fixed-order reduce)
__global__ void reduce_partial_kernel(const float* __restrict__ partial,
                                      float* __restrict__ vbuf) {
  const int i = blockIdx.x * blockDim.x + threadIdx.x;  // over BN*DD
  float s = 0.0f;
#pragma unroll
  for (int c = 0; c < NC; ++c) s += partial[(size_t)c * BN * DD + i];
  vbuf[i] = s;
}

extern "C" void kernel_launch(void* const* d_in, const int* in_sizes, int n_in,
                              void* d_out, int out_size, void* d_ws, size_t ws_size,
                              hipStream_t stream) {
  const float* ds    = (const float*)d_in[0];  // [B,1,D] -> [B,D]
  const float* LF    = (const float*)d_in[1];  // [B,T,D]
  const float* W_dec = (const float*)d_in[2];  // [D,D]
  const float* b_dec = (const float*)d_in[3];  // [D]
  const float* W_enc = (const float*)d_in[4];  // [D,D]
  const float* b_enc = (const float*)d_in[5];  // [D]

  float* out     = (float*)d_out;
  float* attn    = out;                        // [B,T]
  float* context = out + (size_t)BN * TT;      // [B,D]

  // workspace layout (floats), ~3 MB total
  float* ws      = (float*)d_ws;
  float* q       = ws;                                  // 32768
  float* qhat    = q + (size_t)BN * DD;                 // 32768
  float* cb      = qhat + (size_t)BN * DD;              // 64
  float* energy  = cb + 64;                             // 131072
  float* partial = energy + (size_t)BN * TT;            // NC*BN*DD = 524288
  float* vbuf    = partial + (size_t)NC * BN * DD;      // 32768

  // 1) q = ds @ W_dec^T + b_dec        [64x512] (WMMA)
  wmma_gemm_f32<true><<<dim3(DD / 16, BN / 16), 32, 0, stream>>>(
      ds, W_dec, b_dec, q, BN, DD, DD);

  // 2) qhat = q @ W_enc                 [64x512] (WMMA)
  wmma_gemm_f32<false><<<dim3(DD / 16, BN / 16), 32, 0, stream>>>(
      q, W_enc, nullptr, qhat, BN, DD, DD);

  // 3) c[b] = q[b] . b_enc
  dot_bias_kernel<<<BN, 32, 0, stream>>>(q, b_enc, cb);

  // 4) energy[b,t] = LF[b,t,:].qhat[b] + c[b]   (stream LF, 256 MB)
  {
    const int waves = BN * TT / 4;             // one wave per 4 rows
    const int block = 256;                     // 8 waves/block
    energy_kernel<<<waves * 32 / block, block, 0, stream>>>(LF, qhat, cb, energy);
  }

  // 5) attn = softmax(energy) -> d_out
  softmax_kernel<<<BN, 256, 0, stream>>>(energy, attn);

  // 6) v[b,:] = sum_t attn[b,t] * LF[b,t,:]    (stream LF again, 256 MB)
  weighted_sum_kernel<<<dim3(BN, NC), 128, 0, stream>>>(LF, attn, partial);
  reduce_partial_kernel<<<(BN * DD) / 256, 256, 0, stream>>>(partial, vbuf);

  // 7) context = v @ W_enc^T + b_enc    [64x512] (WMMA) -> d_out
  wmma_gemm_f32<true><<<dim3(DD / 16, BN / 16), 32, 0, stream>>>(
      vbuf, W_enc, b_enc, context, BN, DD, DD);
}
